// ARMA_NC_43542378447171
// MI455X (gfx1250) — compile-verified, matrix operand order checked by reading, not run on hardware
//
#include <hip/hip_runtime.h>
#include <hip/hip_bf16.h>

#define NN 50000
#define NE 800000

typedef __attribute__((ext_vector_type(16))) __bf16          v16bf;
typedef __attribute__((ext_vector_type(16))) unsigned short  v16us;
typedef __attribute__((ext_vector_type(8)))  float           v8f;

__device__ __forceinline__ unsigned short f2bf(float f) {
  unsigned u = __float_as_uint(f);
  u += 0x7FFFu + ((u >> 16) & 1u);        // round-to-nearest-even
  return (unsigned short)(u >> 16);
}

// ---------------------------------------------------------------- utilities
__global__ void zero_f32(float* __restrict__ p, int n) {
  int i = blockIdx.x * blockDim.x + threadIdx.x;
  if (i < n) p[i] = 0.f;
}

__global__ void deg_accum(const int* __restrict__ dst, float* __restrict__ deg, int E) {
  int e = blockIdx.x * blockDim.x + threadIdx.x;
  if (e < E) atomicAdd(&deg[dst[e]], 1.0f);
}

__global__ void deg_rsqrt(float* __restrict__ deg, int n) {
  int i = blockIdx.x * blockDim.x + threadIdx.x;
  if (i < n) { float d = deg[i]; deg[i] = (d > 0.f) ? rsqrtf(d) : 0.f; }
}

__global__ void edge_norm(const int* __restrict__ src, const int* __restrict__ dst,
                          const float* __restrict__ dinv, float* __restrict__ nrm, int E) {
  int e = blockIdx.x * blockDim.x + threadIdx.x;
  if (e < E) nrm[e] = dinv[src[e]] * dinv[dst[e]];
}

__global__ void cvt_f32_bf16(const float* __restrict__ in, unsigned short* __restrict__ out, int n) {
  int i = blockIdx.x * blockDim.x + threadIdx.x;
  if (i < n) out[i] = f2bf(in[i]);
}

// --------------------------------------------------- weight / bias builders
// B1 [128,256]: cols 0-127 = v1 (k0|k1), cols 128-255 = w1_init (k0|k1)
__global__ void build_B1(const float* __restrict__ v1, const float* __restrict__ w1i,
                         unsigned short* __restrict__ B) {
  int idx = blockIdx.x * blockDim.x + threadIdx.x;
  if (idx >= 128 * 256) return;
  int r = idx >> 8, c = idx & 255;
  float val;
  if (c < 128) { int k = c >> 6, j = c & 63; val = v1[(k * 128 + r) * 64 + j]; }
  else { int cc = c - 128; int k = cc >> 6, j = cc & 63; val = w1i[(k * 128 + r) * 64 + j]; }
  B[idx] = f2bf(val);
}

// Bw1 [128,128] = blockdiag(w1[0], w1[1]),  w1: [K,64,64]
__global__ void build_Bw1(const float* __restrict__ w1, unsigned short* __restrict__ B) {
  int idx = blockIdx.x * blockDim.x + threadIdx.x;
  if (idx >= 128 * 128) return;
  int r = idx >> 7, c = idx & 127;
  int k = c >> 6, j = c & 63, rk = r >> 6, rj = r & 63;
  float val = (rk == k) ? w1[(k * 64 + rj) * 64 + j] : 0.f;
  B[idx] = f2bf(val);
}

// B2 [64,192]: cols 0-95 = v2 padded 40->48 (k0|k1), cols 96-191 = w2_init padded
__global__ void build_B2(const float* __restrict__ v2, const float* __restrict__ w2i,
                         unsigned short* __restrict__ B) {
  int idx = blockIdx.x * blockDim.x + threadIdx.x;
  if (idx >= 64 * 192) return;
  int r = idx / 192, c = idx - r * 192;
  float val = 0.f;
  if (c < 96) { int k = c / 48, j = c - k * 48; if (j < 40) val = v2[(k * 64 + r) * 40 + j]; }
  else { int cc = c - 96; int k = cc / 48, j = cc - k * 48; if (j < 40) val = w2i[(k * 64 + r) * 40 + j]; }
  B[idx] = f2bf(val);
}

// Bw2 [96,96] = blockdiag(w2[0] pad48, w2[1] pad48),  w2: [K,40,40]
__global__ void build_Bw2(const float* __restrict__ w2, unsigned short* __restrict__ B) {
  int idx = blockIdx.x * blockDim.x + threadIdx.x;
  if (idx >= 96 * 96) return;
  int r = idx / 96, c = idx - r * 96;
  int k = c / 48, j = c - k * 48, rk = r / 48, rj = r - rk * 48;
  float val = (rk == k && rj < 40 && j < 40) ? w2[(k * 40 + rj) * 40 + j] : 0.f;
  B[idx] = f2bf(val);
}

__global__ void build_bias1(const float* __restrict__ b1, float* __restrict__ bias) {
  int i = threadIdx.x;           // 128 threads
  if (i < 128) { int k = i >> 6, j = i & 63; bias[i] = b1[k * 64 + j]; }
}

__global__ void build_bias2(const float* __restrict__ b2, float* __restrict__ bias) {
  int i = threadIdx.x;           // 96 threads
  if (i < 96) { int k = i / 48, j = i - k * 48; bias[i] = (j < 40) ? b2[k * 40 + j] : 0.f; }
}

// ----------------------------------------------------------- WMMA bf16 GEMM
// C[M,Ncols] (f32) = A[M,Kdim=NK*32] (bf16) @ B[Kdim,Ncols] (bf16), row-major.
// NK is a compile-time constant so bfrag[] stays in statically-addressed
// VGPRs (no v_movrels).  One wave owns a 16-wide N tile and streams
// mtiles_per_block 16-row M tiles; inner loop per k-tile is two
// global_load_b128 (A fragment) + one v_wmma_f32_16x16x32_bf16.
template <int NK>
__global__ void gemm_bf16_wmma(const unsigned short* __restrict__ A,
                               const unsigned short* __restrict__ B,
                               float* __restrict__ C,
                               int M, int Ncols,
                               int lda, int ldb, int ldc,
                               int mtiles_per_block) {
  const int lane  = threadIdx.x & 31;
  const int wave  = threadIdx.x >> 5;
  const int ntile = blockIdx.y * (blockDim.x >> 5) + wave;
  const int n0    = ntile << 4;
  if (n0 >= Ncols) return;                       // wave-uniform: EXEC stays all-1s
  const int r  = (lane < 16) ? lane : (lane - 16);
  const int kb = (lane < 16) ? 0 : 8;

  v16bf bfrag[NK];
#pragma unroll
  for (int kt = 0; kt < NK; ++kt) {
    const unsigned short* Bp = B + (size_t)(kt * 32 + kb) * ldb + (n0 + r);
    v16us t;
#pragma unroll
    for (int j = 0; j < 8; ++j) t[j]     = Bp[(size_t)j * ldb];
#pragma unroll
    for (int j = 0; j < 8; ++j) t[8 + j] = Bp[(size_t)(16 + j) * ldb];
    bfrag[kt] = __builtin_bit_cast(v16bf, t);
  }

  const int mt0 = blockIdx.x * mtiles_per_block;
  for (int mi = 0; mi < mtiles_per_block; ++mi) {
    const int m0 = (mt0 + mi) << 4;
    if (m0 >= M) break;                          // uniform per wave
    const unsigned short* Arow = A + (size_t)(m0 + r) * lda + kb;
    __builtin_prefetch(Arow + (size_t)16 * lda, 0, 0);   // next M tile
    v8f acc = {};
#pragma unroll
    for (int kt = 0; kt < NK; ++kt) {
      const unsigned short* Ap = Arow + kt * 32;
      v16us t;
#pragma unroll
      for (int j = 0; j < 8; ++j) t[j]     = Ap[j];
#pragma unroll
      for (int j = 0; j < 8; ++j) t[8 + j] = Ap[16 + j];
      v16bf afrag = __builtin_bit_cast(v16bf, t);
      acc = __builtin_amdgcn_wmma_f32_16x16x32_bf16(
          false, afrag, false, bfrag[kt], (short)0, acc, false, false);
    }
    // C/D layout: lane<16 -> (M=v, N=lane); lane>=16 -> (M=v+8, N=lane-16)
    float* Cp = C + (size_t)(m0 + ((lane < 16) ? 0 : 8)) * ldc + (n0 + r);
#pragma unroll
    for (int v = 0; v < 8; ++v) Cp[(size_t)v * ldc] = acc[v];
  }
}

// ----------------------------------------------------------- sparse scatter
// agg[dst] += norm_e * H[src]   (edge- and 4-column-parallel, f32 atomics; L2 resident)
__global__ void spmm_scatter(const int* __restrict__ src, const int* __restrict__ dst,
                             const float* __restrict__ nrm,
                             const float* __restrict__ H, int ldh,
                             float* __restrict__ AGG, int ldagg,
                             int E, int chunks) {
  int tid = blockIdx.x * blockDim.x + threadIdx.x;
  int total = E * chunks;
  if (tid >= total) return;
  int e  = tid / chunks;
  int c4 = (tid - e * chunks) << 2;
  int s = src[e], d = dst[e];
  float w = nrm[e];
  const float4 h = *(const float4*)(H + (size_t)s * ldh + c4);
  float* ap = AGG + (size_t)d * ldagg + c4;
  atomicAdd(ap + 0, w * h.x);
  atomicAdd(ap + 1, w * h.y);
  atomicAdd(ap + 2, w * h.z);
  atomicAdd(ap + 3, w * h.w);
}

// out = (relu?)(agg + root + bias); writes f32 (may alias agg) and bf16
__global__ void combine(const float* __restrict__ agg,
                        const float* __restrict__ root, int ldroot,
                        const float* __restrict__ bias,
                        float* __restrict__ outF, unsigned short* __restrict__ outB,
                        int total, int cols, int do_relu) {
  int idx = blockIdx.x * blockDim.x + threadIdx.x;
  if (idx >= total) return;
  int n = idx / cols, c = idx - n * cols;
  float v = agg[idx] + root[(size_t)n * ldroot + c] + bias[c];
  if (do_relu) v = fmaxf(v, 0.f);
  outF[idx] = v;
  outB[idx] = f2bf(v);
}

// hmid[n,c] = bf16(relu(0.5*(in[n,c]+in[n,64+c]))),  in: [N,128]
__global__ void mean_relu_cvt(const float* __restrict__ in, unsigned short* __restrict__ out, int total) {
  int idx = blockIdx.x * blockDim.x + threadIdx.x;
  if (idx >= total) return;
  int n = idx >> 6, c = idx & 63;
  float v = 0.5f * (in[(size_t)n * 128 + c] + in[(size_t)n * 128 + 64 + c]);
  out[idx] = f2bf(fmaxf(v, 0.f));
}

// o = log_softmax(0.5*(in[:,0:40] + in[:,48:88])),  in: [N,96] -> out: [N,40]
__global__ void final_logsoftmax(const float* __restrict__ in, float* __restrict__ out, int Nn) {
  int r = blockIdx.x * blockDim.x + threadIdx.x;
  if (r >= Nn) return;
  const float* b = in + (size_t)r * 96;
  float vals[40];
  float m = -3.0e38f;
  for (int c = 0; c < 40; ++c) {
    float v = 0.5f * (b[c] + b[48 + c]);
    vals[c] = v;
    m = fmaxf(m, v);
  }
  float s = 0.f;
  for (int c = 0; c < 40; ++c) s += __expf(vals[c] - m);
  float ls = __logf(s);
  float* o = out + (size_t)r * 40;
  for (int c = 0; c < 40; ++c) o[c] = vals[c] - m - ls;
}

// ---------------------------------------------------------------- launcher
extern "C" void kernel_launch(void* const* d_in, const int* in_sizes, int n_in,
                              void* d_out, int out_size, void* d_ws, size_t ws_size,
                              hipStream_t stream) {
  (void)in_sizes; (void)n_in; (void)out_size; (void)ws_size;
  const float* x       = (const float*)d_in[0];
  const int*   ei      = (const int*)d_in[1];
  const int*   src     = ei;
  const int*   dst     = ei + NE;
  const float* w1_init = (const float*)d_in[2];
  const float* w1      = (const float*)d_in[3];
  const float* v1      = (const float*)d_in[4];
  const float* b1      = (const float*)d_in[5];
  const float* w2_init = (const float*)d_in[6];
  const float* w2      = (const float*)d_in[7];
  const float* v2      = (const float*)d_in[8];
  const float* b2      = (const float*)d_in[9];
  float* out = (float*)d_out;

  char* ws = (char*)d_ws;
  size_t off = 0;
  auto alloc = [&](size_t bytes) -> void* {
    void* p = ws + off;
    off += (bytes + 255) & ~(size_t)255;
    return p;
  };

  float*          dinv  = (float*)alloc((size_t)NN * 4);
  float*          enrm  = (float*)alloc((size_t)NE * 4);
  unsigned short* xbf   = (unsigned short*)alloc((size_t)NN * 128 * 2);
  unsigned short* B1    = (unsigned short*)alloc(128 * 256 * 2);
  unsigned short* Bw1   = (unsigned short*)alloc(128 * 128 * 2);
  unsigned short* B2    = (unsigned short*)alloc(64 * 192 * 2);
  unsigned short* Bw2   = (unsigned short*)alloc(96 * 96 * 2);
  float*          bias1 = (float*)alloc(128 * 4);
  float*          bias2 = (float*)alloc(96 * 4);
  float*          Cbuf  = (float*)alloc((size_t)NN * 256 * 4);  // C1 / C2 (aliased)
  float*          aggb  = (float*)alloc((size_t)NN * 128 * 4);  // agg1 / agg2 / final
  unsigned short* obf   = (unsigned short*)alloc((size_t)NN * 128 * 2);
  unsigned short* hmid  = (unsigned short*)alloc((size_t)NN * 64 * 2);

  const int TB = 256;
  auto blks = [](int n) { return (n + 255) / 256; };

  // gcn_norm
  zero_f32<<<blks(NN), TB, 0, stream>>>(dinv, NN);
  deg_accum<<<blks(NE), TB, 0, stream>>>(dst, dinv, NE);
  deg_rsqrt<<<blks(NN), TB, 0, stream>>>(dinv, NN);
  edge_norm<<<blks(NE), TB, 0, stream>>>(src, dst, dinv, enrm, NE);

  // bf16 prep
  cvt_f32_bf16<<<blks(NN * 128), TB, 0, stream>>>(x, xbf, NN * 128);
  build_B1 <<<blks(128 * 256), TB, 0, stream>>>(v1, w1_init, B1);
  build_Bw1<<<blks(128 * 128), TB, 0, stream>>>(w1, Bw1);
  build_B2 <<<blks(64 * 192),  TB, 0, stream>>>(v2, w2_init, B2);
  build_Bw2<<<blks(96 * 96),   TB, 0, stream>>>(w2, Bw2);
  build_bias1<<<1, 128, 0, stream>>>(b1, bias1);
  build_bias2<<<1, 96, 0, stream>>>(b2, bias2);

  const int MT = 8;  // M tiles per block
  auto ggrid = [&](int M, int Nc) { return dim3((M / 16 + MT - 1) / MT, (Nc / 16 + 3) / 4); };

  // ---------------- Layer 1 (HID=64, K=2 -> 128 cols; ReLU) ----------------
  // C1[N,256] = [ root(k0|k1) | h0(k0|k1) ]
  gemm_bf16_wmma<4><<<ggrid(NN, 256), 128, 0, stream>>>(xbf, B1, Cbuf, NN, 256, 128, 256, 256, MT);
  zero_f32<<<blks(NN * 128), TB, 0, stream>>>(aggb, NN * 128);
  spmm_scatter<<<blks(NE * 32), TB, 0, stream>>>(src, dst, enrm, Cbuf + 128, 256, aggb, 128, NE, 32);
  combine<<<blks(NN * 128), TB, 0, stream>>>(aggb, Cbuf, 256, bias1, aggb, obf, NN * 128, 128, 1);
  // h = out @ blockdiag(w1)  -> write into h-region of C1
  gemm_bf16_wmma<4><<<ggrid(NN, 128), 128, 0, stream>>>(obf, Bw1, Cbuf + 128, NN, 128, 128, 128, 256, MT);
  zero_f32<<<blks(NN * 128), TB, 0, stream>>>(aggb, NN * 128);
  spmm_scatter<<<blks(NE * 32), TB, 0, stream>>>(src, dst, enrm, Cbuf + 128, 256, aggb, 128, NE, 32);
  combine<<<blks(NN * 128), TB, 0, stream>>>(aggb, Cbuf, 256, bias1, aggb, obf, NN * 128, 128, 1);
  // mean over stacks + wrapper ReLU -> bf16 [N,64]
  mean_relu_cvt<<<blks(NN * 64), TB, 0, stream>>>(aggb, hmid, NN * 64);

  // ---------------- Layer 2 (NCLS=40 padded to 48 -> 96 cols; identity) ----
  // C2[N,192] = [ root2(k0|k1) | h0_2(k0|k1) ]
  gemm_bf16_wmma<2><<<ggrid(NN, 192), 128, 0, stream>>>(hmid, B2, Cbuf, NN, 192, 64, 192, 192, MT);
  zero_f32<<<blks(NN * 96), TB, 0, stream>>>(aggb, NN * 96);
  spmm_scatter<<<blks(NE * 24), TB, 0, stream>>>(src, dst, enrm, Cbuf + 96, 192, aggb, 96, NE, 24);
  combine<<<blks(NN * 96), TB, 0, stream>>>(aggb, Cbuf, 192, bias2, aggb, obf, NN * 96, 96, 0);
  gemm_bf16_wmma<3><<<ggrid(NN, 96), 128, 0, stream>>>(obf, Bw2, Cbuf + 96, NN, 96, 96, 96, 192, MT);
  zero_f32<<<blks(NN * 96), TB, 0, stream>>>(aggb, NN * 96);
  spmm_scatter<<<blks(NE * 24), TB, 0, stream>>>(src, dst, enrm, Cbuf + 96, 192, aggb, 96, NE, 24);
  combine<<<blks(NN * 96), TB, 0, stream>>>(aggb, Cbuf, 192, bias2, aggb, obf, NN * 96, 96, 0);

  // mean over stacks + log_softmax -> d_out [N,40]
  final_logsoftmax<<<blks(NN), TB, 0, stream>>>(aggb, out, NN);
}
